// CausalSelfAttention_70282844832424
// MI455X (gfx1250) — compile-verified, matrix operand order checked by reading, not run on hardware
//
#include <hip/hip_runtime.h>
#include <hip/hip_bf16.h>

// ---------------------------------------------------------------------------
// Causal self-attention with polymax, MI455X (gfx1250, wave32, WMMA).
// Pipeline: [qkv GEMM f32->f16] -> [fused flash attention + polymax] -> [proj GEMM]
// All matrix math: v_wmma_f32_16x16x32_f16 (f16 operands, f32 accumulate).
// ---------------------------------------------------------------------------

typedef _Float16 h16;
typedef __attribute__((ext_vector_type(16))) _Float16 v16h;
typedef __attribute__((ext_vector_type(8)))  _Float16 v8h;
typedef __attribute__((ext_vector_type(4)))  _Float16 v4h;
typedef __attribute__((ext_vector_type(8)))  float    v8f;
typedef __attribute__((ext_vector_type(4)))  float    v4f;

#define BDIM 2
#define TDIM 2048
#define CDIM 1024
#define NH   16
#define HD   64

static __device__ __forceinline__ v8f wmma16(v16h a, v16h b, v8f c) {
  // (neg_a, A, neg_b, B, c_mod, C, reuse_a, reuse_b)
  return __builtin_amdgcn_wmma_f32_16x16x32_f16(false, a, false, b, (short)0, c,
                                                false, false);
}
static __device__ __forceinline__ v16h joinh(v8h lo, v8h hi) {
  return __builtin_shufflevector(lo, hi, 0, 1, 2, 3, 4, 5, 6, 7,
                                         8, 9, 10, 11, 12, 13, 14, 15);
}

// ---------------------------------------------------------------------------
// Kernel 1: qkv = x @ W_attn, converting f32->f16 on the fly, scattering the
// result to head-major f16 q/k/v tensors [B,H,T,hd].
// Block tile 128x128, BK=32. 8 waves: wave tile 32(M) x 64(N).
// ---------------------------------------------------------------------------
__global__ __launch_bounds__(256)
void qkv_gemm_kernel(const float* __restrict__ x, const float* __restrict__ Wa,
                     h16* __restrict__ qf, h16* __restrict__ kf,
                     h16* __restrict__ vf) {
  const int N = 3 * CDIM;  // 3072
  const int K = CDIM;      // 1024
  __shared__ __align__(16) h16 As[128][40];  // [m][k], padded
  __shared__ __align__(16) h16 Bs[128][40];  // [n][k] (transposed), padded

  const int tid  = threadIdx.x;
  const int lane = tid & 31;
  const int wave = tid >> 5;
  const int wm   = wave >> 1;  // 0..3
  const int wn   = wave & 1;   // 0..1
  const int m0   = (blockIdx.x / 24) * 128;
  const int n0   = (blockIdx.x % 24) * 128;

  const int r  = lane & 15;  // fragment row / col
  const int kh = lane >> 4;  // K-half selector (A/B 16-bit fragment layout)

  v8f z = {};
  v8f acc[2][4];
#pragma unroll
  for (int i = 0; i < 2; ++i)
#pragma unroll
    for (int j = 0; j < 4; ++j) acc[i][j] = z;

  const int lr = tid >> 3;  // 0..31
  const int ls = tid & 7;   // 0..7

  for (int k0 = 0; k0 < K; k0 += 32) {
    // stage x tile (128x32 f32) -> As f16
#pragma unroll
    for (int p = 0; p < 4; ++p) {
      int row = p * 32 + lr;
      v4f xv = *(const v4f*)(x + (size_t)(m0 + row) * K + k0 + ls * 4);
      v4h hv;
      hv[0] = (h16)xv[0]; hv[1] = (h16)xv[1];
      hv[2] = (h16)xv[2]; hv[3] = (h16)xv[3];
      *(v4h*)&As[row][ls * 4] = hv;
    }
    // stage W tile (32x128 f32) -> Bs f16, transposed to [n][k]
#pragma unroll
    for (int p = 0; p < 4; ++p) {
      int ncol = (p * 8 + ls) * 4;
      v4f wv = *(const v4f*)(Wa + (size_t)(k0 + lr) * N + n0 + ncol);
#pragma unroll
      for (int e = 0; e < 4; ++e) Bs[ncol + e][lr] = (h16)wv[e];
    }
    if (k0 + 32 < K) {  // global_prefetch_b8 on next K tile
      __builtin_prefetch(x + (size_t)(m0 + lr) * K + k0 + 32, 0, 1);
      __builtin_prefetch(Wa + (size_t)(k0 + 32 + lr) * N + n0, 0, 1);
    }
    __syncthreads();
#pragma unroll
    for (int mi = 0; mi < 2; ++mi) {
      const h16* ap = &As[wm * 32 + mi * 16 + r][0];
      v16h af = joinh(*(const v8h*)(ap + kh * 8),
                      *(const v8h*)(ap + 16 + kh * 8));
#pragma unroll
      for (int ni = 0; ni < 4; ++ni) {
        const h16* bp = &Bs[wn * 64 + ni * 16 + r][0];
        v16h bf = joinh(*(const v8h*)(bp + kh * 8),
                        *(const v8h*)(bp + 16 + kh * 8));
        acc[mi][ni] = wmma16(af, bf, acc[mi][ni]);
      }
    }
    __syncthreads();
  }

  // C-layout: row = kh*8 + i, col = r. Scatter to q/k/v [B,H,T,hd] as f16.
#pragma unroll
  for (int mi = 0; mi < 2; ++mi) {
#pragma unroll
    for (int ni = 0; ni < 4; ++ni) {
#pragma unroll
      for (int i = 0; i < 8; ++i) {
        int gm  = m0 + wm * 32 + mi * 16 + kh * 8 + i;
        int gn  = n0 + wn * 64 + ni * 16 + r;
        int b   = gm >> 11, t = gm & (TDIM - 1);
        int wch = gn >> 10, cc = gn & (CDIM - 1);
        int hh  = cc >> 6,  d  = cc & (HD - 1);
        h16* dst = (wch == 0) ? qf : (wch == 1) ? kf : vf;
        dst[(((size_t)b * NH + hh) * TDIM + t) * HD + d] = (h16)acc[mi][ni][i];
      }
    }
  }
}

// ---------------------------------------------------------------------------
// Kernel 2: fused flash-style attention with polymax.
// Block = (b, h, 128 query rows); each of 8 waves owns 16 query rows.
// Per 128-key tile: S = qK^T (wmma) -> mask+polymax in f32 -> P f16 via LDS
// -> Y += P V (wmma). Causal loop stops at the diagonal tile.
// ---------------------------------------------------------------------------
__global__ __launch_bounds__(256)
void attn_kernel(const h16* __restrict__ qf, const h16* __restrict__ kf,
                 const h16* __restrict__ vf, h16* __restrict__ yf) {
  __shared__ __align__(16) h16 Ks[128][72];      // [key][d]
  __shared__ __align__(16) h16 Vt[64][136];      // [d][key] (transposed)
  __shared__ __align__(16) h16 Ps[8][16][136];   // per-wave P tile [row][key]

  const int tid  = threadIdx.x;
  const int lane = tid & 31;
  const int wave = tid >> 5;
  const int r    = lane & 15;
  const int kh   = lane >> 4;

  const int qt    = blockIdx.x & 15;   // query tile (T/128 = 16)
  const int bh    = blockIdx.x >> 4;   // b*NH + h
  const int b     = bh >> 4;
  const int hh    = bh & 15;
  const int qbase = qt * 128;

  const h16* qhp = qf + (size_t)bh * TDIM * HD;
  const h16* khp = kf + (size_t)bh * TDIM * HD;
  const h16* vhp = vf + (size_t)bh * TDIM * HD;

  // Q fragments (A-layout, 16 rows x 64 dims) live in registers throughout.
  v16h Aq[2];
#pragma unroll
  for (int kk = 0; kk < 2; ++kk) {
    const h16* qp = qhp + (size_t)(qbase + wave * 16 + r) * HD + kk * 32;
    Aq[kk] = joinh(*(const v8h*)(qp + kh * 8), *(const v8h*)(qp + 16 + kh * 8));
  }

  v8f z = {};
  v8f Y[4];
#pragma unroll
  for (int i = 0; i < 4; ++i) Y[i] = z;

  const int lr = tid >> 1;  // 0..127 (key row for staging)
  const int ls = tid & 1;   // 0..1   (32-dim half)
  const float scale = 0.125f;  // 1/sqrt(64)

  for (int kt = 0; kt <= qt; ++kt) {
    const int kb = kt * 128;
    // stage K tile [key][d]
#pragma unroll
    for (int j = 0; j < 4; ++j) {
      v8h kv = *(const v8h*)(khp + (size_t)(kb + lr) * HD + ls * 32 + j * 8);
      *(v8h*)&Ks[lr][ls * 32 + j * 8] = kv;
    }
    // stage V tile transposed [d][key]
#pragma unroll
    for (int j = 0; j < 4; ++j) {
      v8h vv = *(const v8h*)(vhp + (size_t)(kb + lr) * HD + ls * 32 + j * 8);
#pragma unroll
      for (int e = 0; e < 8; ++e) Vt[ls * 32 + j * 8 + e][lr] = vv[e];
    }
    __syncthreads();

    // S = q k^T over hd=64 (2 wmma K-steps), 8 key sub-tiles of 16
#pragma unroll
    for (int nn = 0; nn < 8; ++nn) {
      v8f S = z;
#pragma unroll
      for (int kk = 0; kk < 2; ++kk) {
        const h16* bp = &Ks[nn * 16 + r][kk * 32];
        v16h bf = joinh(*(const v8h*)(bp + kh * 8),
                        *(const v8h*)(bp + 16 + kh * 8));
        S = wmma16(Aq[kk], bf, S);
      }
      // causal mask + polymax (piecewise), then P -> per-wave LDS tile
#pragma unroll
      for (int i = 0; i < 8; ++i) {
        int rr = kh * 8 + i;                 // C-layout row
        int qg = qbase + wave * 16 + rr;     // global query index
        int kg = kb + nn * 16 + r;           // global key index
        float xs = S[i] * scale;
        float p;
        if (kg > qg)            p = 0.f;                       // masked -> flat piece
        else if (xs > 0.f)      p = (xs * xs + 1.f) * 0.001f;  // (x^2 + 1)/1000
        else if (xs >= -100.f)  p = (1.f - 0.01f * xs) * 0.001f; // (m*x + 1)/1000
        else                    p = 0.f;
        Ps[wave][rr][nn * 16 + r] = (h16)p;
      }
    }

    // Y += P @ V  (K dim = 128 keys -> 4 wmma K-steps, 4 dim sub-tiles)
#pragma unroll
    for (int kk2 = 0; kk2 < 4; ++kk2) {
      const h16* pp = &Ps[wave][r][kk2 * 32];
      v16h af = joinh(*(const v8h*)(pp + kh * 8),
                      *(const v8h*)(pp + 16 + kh * 8));
#pragma unroll
      for (int nd = 0; nd < 4; ++nd) {
        const h16* bp = &Vt[nd * 16 + r][kk2 * 32];
        v16h bf = joinh(*(const v8h*)(bp + kh * 8),
                        *(const v8h*)(bp + 16 + kh * 8));
        Y[nd] = wmma16(af, bf, Y[nd]);
      }
    }
    __syncthreads();
  }

  // write Y -> yf [B,T,C] f16
#pragma unroll
  for (int nd = 0; nd < 4; ++nd) {
#pragma unroll
    for (int i = 0; i < 8; ++i) {
      int t = qbase + wave * 16 + kh * 8 + i;
      int d = nd * 16 + r;
      yf[((size_t)b * TDIM + t) * CDIM + hh * HD + d] = (h16)Y[nd][i];
    }
  }
}

// ---------------------------------------------------------------------------
// Kernel 3: out = y @ W_proj. y is f16 in workspace, W_proj converted f32->f16
// during staging, output f32 to d_out. Same tiling as kernel 1.
// ---------------------------------------------------------------------------
__global__ __launch_bounds__(256)
void proj_gemm_kernel(const h16* __restrict__ yf, const float* __restrict__ Wp,
                      float* __restrict__ out) {
  const int N = CDIM, K = CDIM;
  __shared__ __align__(16) h16 As[128][40];
  __shared__ __align__(16) h16 Bs[128][40];

  const int tid  = threadIdx.x;
  const int lane = tid & 31;
  const int wave = tid >> 5;
  const int wm   = wave >> 1;
  const int wn   = wave & 1;
  const int m0   = (blockIdx.x >> 3) * 128;
  const int n0   = (blockIdx.x & 7) * 128;

  const int r  = lane & 15;
  const int kh = lane >> 4;

  v8f z = {};
  v8f acc[2][4];
#pragma unroll
  for (int i = 0; i < 2; ++i)
#pragma unroll
    for (int j = 0; j < 4; ++j) acc[i][j] = z;

  const int ar = tid >> 1;  // 0..127 (A rows)
  const int as = tid & 1;
  const int lr = tid >> 3;  // 0..31  (B k-rows)
  const int ls = tid & 7;

  for (int k0 = 0; k0 < K; k0 += 32) {
    // stage y tile (already f16) -> As
#pragma unroll
    for (int j = 0; j < 2; ++j) {
      v8h av = *(const v8h*)(yf + (size_t)(m0 + ar) * K + k0 + as * 16 + j * 8);
      *(v8h*)&As[ar][as * 16 + j * 8] = av;
    }
    // stage W_proj tile f32 -> Bs f16 transposed
#pragma unroll
    for (int p = 0; p < 4; ++p) {
      int ncol = (p * 8 + ls) * 4;
      v4f wv = *(const v4f*)(Wp + (size_t)(k0 + lr) * N + n0 + ncol);
#pragma unroll
      for (int e = 0; e < 4; ++e) Bs[ncol + e][lr] = (h16)wv[e];
    }
    if (k0 + 32 < K) {
      __builtin_prefetch(yf + (size_t)(m0 + ar) * K + k0 + 32, 0, 1);
      __builtin_prefetch(Wp + (size_t)(k0 + 32 + lr) * N + n0, 0, 1);
    }
    __syncthreads();
#pragma unroll
    for (int mi = 0; mi < 2; ++mi) {
      const h16* ap = &As[wm * 32 + mi * 16 + r][0];
      v16h af = joinh(*(const v8h*)(ap + kh * 8),
                      *(const v8h*)(ap + 16 + kh * 8));
#pragma unroll
      for (int ni = 0; ni < 4; ++ni) {
        const h16* bp = &Bs[wn * 64 + ni * 16 + r][0];
        v16h bf = joinh(*(const v8h*)(bp + kh * 8),
                        *(const v8h*)(bp + 16 + kh * 8));
        acc[mi][ni] = wmma16(af, bf, acc[mi][ni]);
      }
    }
    __syncthreads();
  }

#pragma unroll
  for (int mi = 0; mi < 2; ++mi) {
#pragma unroll
    for (int ni = 0; ni < 4; ++ni) {
#pragma unroll
      for (int i = 0; i < 8; ++i) {
        int gm = m0 + wm * 32 + mi * 16 + kh * 8 + i;
        int gn = n0 + wn * 64 + ni * 16 + r;
        out[(size_t)gm * N + gn] = acc[mi][ni][i];
      }
    }
  }
}

// ---------------------------------------------------------------------------
extern "C" void kernel_launch(void* const* d_in, const int* in_sizes, int n_in,
                              void* d_out, int out_size, void* d_ws,
                              size_t ws_size, hipStream_t stream) {
  (void)in_sizes; (void)n_in; (void)out_size; (void)ws_size;

  const float* x  = (const float*)d_in[0];  // [2,2048,1024]
  const float* Wa = (const float*)d_in[1];  // [1024,3072]
  const float* Wp = (const float*)d_in[2];  // [1024,1024]
  float* out = (float*)d_out;               // [2,2048,1024] f32

  // f16 workspace: q,k,v head-major [B,H,T,hd] + y [B,T,C]  (32 MB total)
  h16* ws = (h16*)d_ws;
  const size_t HTD = (size_t)BDIM * NH * TDIM * HD;  // 4,194,304
  h16* qf = ws;
  h16* kf = qf + HTD;
  h16* vf = kf + HTD;
  h16* yf = vf + HTD;

  qkv_gemm_kernel<<<dim3(32 * 24), 256, 0, stream>>>(x, Wa, qf, kf, vf);
  attn_kernel<<<dim3(BDIM * NH * (TDIM / 128)), 256, 0, stream>>>(qf, kf, vf, yf);
  proj_gemm_kernel<<<dim3(32 * 8), 256, 0, stream>>>(yf, Wp, out);
}
